// RNN_Encoder_14577119003417
// MI455X (gfx1250) — matrix operand order repaired and checked by reading, not compiled, not run on hardware
//
#include <hip/hip_runtime.h>
#include <hip/hip_bf16.h>
#include <stdint.h>

// Problem dims
#define B_    128
#define T_    512
#define E_    256
#define H_    128
#define FOURH 512

typedef __attribute__((ext_vector_type(16))) __bf16 v16bf;
typedef __attribute__((ext_vector_type(8)))  __bf16 v8bf;
typedef __attribute__((ext_vector_type(8)))  float  v8f;

// Hardware tanh on gfx1250 (v_tanh_f32, TRANS pipe -> co-executes with WMMA)
#if __has_builtin(__builtin_amdgcn_tanhf)
__device__ __forceinline__ float fast_tanh(float x) { return __builtin_amdgcn_tanhf(x); }
#else
__device__ __forceinline__ float fast_tanh(float x) { return tanhf(x); }
#endif
// sigmoid via tanh identity: sigma(x) = 0.5 + 0.5*tanh(x/2)
__device__ __forceinline__ float fast_sigmoid(float x) {
    return __builtin_fmaf(0.5f, fast_tanh(0.5f * x), 0.5f);
}

__device__ __forceinline__ uint16_t f32_to_bf16(float x) {
    uint32_t u = __float_as_uint(x);
    uint32_t r = u + 0x7FFFu + ((u >> 16) & 1u);   // round-to-nearest-even
    return (uint16_t)(r >> 16);
}

// ---- WMMA fragment loaders (16x16x32 bf16; contiguous k = 16h + s, same for A and B) ----
// A matrix 16xK, row-major, row m = lane&15.
// lanes 0-15: elems 0..15 = K[k0..k0+15]
// lanes16-31: elems 0..15 = K[k0+16..k0+31]
__device__ __forceinline__ v16bf load_frag_a(const uint16_t* base, int ld, int lane, int k0) {
    int m   = lane & 15;
    int sub = (lane >> 4) << 4;            // 0 or 16
    const uint16_t* p = base + (size_t)m * ld + k0 + sub;
    return *(const v16bf*)p;
}

// B matrix Kx16 loaded from pre-transposed weights Wt[N][K] (row-major in N).
// lane&15 -> column n; lanes 0-15 hold K[k0..k0+15], lanes 16-31 hold K[k0+16..k0+31].
__device__ __forceinline__ v16bf load_frag_b(const uint16_t* Wt, int ld, int lane, int n0, int k0) {
    int n  = n0 + (lane & 15);
    int kk = k0 + ((lane >> 4) << 4);
    return *(const v16bf*)(Wt + (size_t)n * ld + kk);
}

// ---- Kernel 0: transpose + convert fp32 weights [K][N] -> bf16 Wt[N][K] ----
__global__ void transpose_cvt(const float* __restrict__ W, uint16_t* __restrict__ Wt,
                              int K, int N) {
    int idx = blockIdx.x * blockDim.x + threadIdx.x;
    if (idx >= K * N) return;
    int k = idx % K;
    int n = idx / K;
    Wt[idx] = f32_to_bf16(W[(size_t)k * N + n]);
}

// ---- Kernel 1: embedding gather (time-major) -> bf16 xt[t*B+b][E] ----
__global__ void gather_embed(const long long* __restrict__ tokens,
                             const float* __restrict__ emb,
                             uint16_t* __restrict__ xt) {
    int r = blockIdx.x;               // r = t*B + b
    int t = r / B_;
    int b = r % B_;
    long long tok = tokens[(long long)b * T_ + t];
    xt[(size_t)r * E_ + threadIdx.x] =
        f32_to_bf16(emb[(size_t)tok * E_ + threadIdx.x]);
}

// ---- Kernel 2: input projection GEMM  xw[dir] = xt @ W[dir] + b[dir]
// Each wave computes a 16x64 strip (4 N-tiles), reusing each A fragment 4x.
// Output stored SWIZZLED in WMMA C-fragment order: xw[tile][lane][8] contiguous,
// tile = tm*32 + tn_idx, so the recurrence reads one v8f per gate per lane.
__global__ __launch_bounds__(256) void inproj_gemm(
    const uint16_t* __restrict__ xt,
    const uint16_t* __restrict__ Wt_fw, const uint16_t* __restrict__ Wt_bw,
    const float* __restrict__ b_fw, const float* __restrict__ b_bw,
    float* __restrict__ xw_fw, float* __restrict__ xw_bw)
{
    const int lane = threadIdx.x & 31;
    const int wave = threadIdx.x >> 5;
    const int gid  = blockIdx.x * 8 + wave;       // strip id
    const int tm   = gid >> 3;                    // 0..4095 (row tile)
    const int tg   = gid & 7;                     // group of 4 N-tiles
    const int dir  = blockIdx.y;
    const uint16_t* Wt  = dir ? Wt_bw : Wt_fw;
    const float*   bias = dir ? b_bw  : b_fw;
    float*           xw = dir ? xw_bw : xw_fw;

    const uint16_t* Arow = xt + (size_t)tm * 16 * E_;
    v8f acc[4] = {{}, {}, {}, {}};
#pragma unroll
    for (int kc = 0; kc < E_ / 32; ++kc) {
        v16bf a = load_frag_a(Arow, E_, lane, kc * 32);
#pragma unroll
        for (int j = 0; j < 4; ++j) {
            v16bf b = load_frag_b(Wt, E_, lane, (tg * 4 + j) * 16, kc * 32);
            acc[j] = __builtin_amdgcn_wmma_f32_16x16x32_bf16(false, a, false, b,
                                                             (short)0, acc[j],
                                                             false, false);
        }
    }
#pragma unroll
    for (int j = 0; j < 4; ++j) {
        int   tn_idx = tg * 4 + j;
        float bv     = bias[tn_idx * 16 + (lane & 15)];   // bias folded here
        v8f r;
#pragma unroll
        for (int v = 0; v < 8; ++v) r[v] = acc[j][v] + bv;
        *(v8f*)(xw + ((size_t)tm * 32 + tn_idx) * 256 + lane * 8) = r;
    }
}

// ---- Kernel 3: sequential LSTM recurrence.
// grid = (8 batch-tiles, 2 directions), 256 threads = 8 waves.
// Wave w owns output columns [w*16, w*16+16) of every gate -> gate math is wave-local.
// U fragments live in registers for the whole loop; h (bf16) is DOUBLE-BUFFERED
// in LDS so only ONE barrier per step is needed. z-init comes from the swizzled
// xw (one v8f load per gate); next step's tiles are prefetched.
__global__ __launch_bounds__(256) void lstm_recurrence(
    const float* __restrict__ xw_fw, const float* __restrict__ xw_bw,
    const uint16_t* __restrict__ Ut_fw, const uint16_t* __restrict__ Ut_bw,
    float* __restrict__ out, float* __restrict__ states)
{
    __shared__ uint16_t h_lds[2][16 * H_];        // 2 x 4 KB double buffer

    const int lane = threadIdx.x & 31;
    const int w    = threadIdx.x >> 5;            // wave id 0..7
    const int bt   = blockIdx.x;                  // batch tile (16 rows)
    const int dir  = blockIdx.y;

    const float*    xw = dir ? xw_bw : xw_fw;
    const uint16_t* Ut = dir ? Ut_bw : Ut_fw;

    const int hcol = w * 16 + (lane & 15);        // column within H
    const int moff = 8 * (lane >> 4);             // C-layout row offset

    // Preload recurrent weights into registers: 4 gates x 4 k-chunks (128 VGPRs/lane)
    v16bf ufrag[4][4];
#pragma unroll
    for (int g = 0; g < 4; ++g)
#pragma unroll
        for (int kc = 0; kc < 4; ++kc)
            ufrag[g][kc] = load_frag_b(Ut, H_, lane, g * H_ + w * 16, kc * 32);

    // zero initial h buffers; zero c in registers
    for (int i = threadIdx.x; i < 2 * 16 * H_; i += 256) h_lds[0][i] = 0;
    float cst[8], hlast[8];
#pragma unroll
    for (int v = 0; v < 8; ++v) { cst[v] = 0.f; hlast[v] = 0.f; }
    __syncthreads();

    for (int step = 0; step < T_; ++step) {
        const int t   = dir ? (T_ - 1 - step) : step;
        const int cur = step & 1;
        const int nxt = cur ^ 1;

        // z init: one v8f (32B) fragment-order load per gate (bias already folded)
        const float* xwbase = xw + ((size_t)(t * 8 + bt) * 32) * 256 + lane * 8;
        v8f z[4];
#pragma unroll
        for (int g = 0; g < 4; ++g)
            z[g] = *(const v8f*)(xwbase + (size_t)(g * 8 + w) * 256);

        // prefetch next timestep's gate tiles (overlaps WMMA + gate math)
        if (step + 1 < T_) {
            const int t2 = dir ? (t - 1) : (t + 1);
            const float* pf = xw + ((size_t)(t2 * 8 + bt) * 32) * 256 + lane * 8;
#pragma unroll
            for (int g = 0; g < 4; ++g)
                __builtin_prefetch(pf + (size_t)(g * 8 + w) * 256, 0, 0);
        }

        // read h (all columns) from current LDS buffer into A fragments
        v16bf afrag[4];
#pragma unroll
        for (int kc = 0; kc < 4; ++kc)
            afrag[kc] = load_frag_a(h_lds[cur], H_, lane, kc * 32);

        // z += h @ U  (one 16x16 tile per gate per wave; 4 independent chains)
#pragma unroll
        for (int g = 0; g < 4; ++g) {
#pragma unroll
            for (int kc = 0; kc < 4; ++kc)
                z[g] = __builtin_amdgcn_wmma_f32_16x16x32_bf16(false, afrag[kc], false,
                                                               ufrag[g][kc], (short)0,
                                                               z[g], false, false);
        }

        // gate nonlinearities + state update (wave-local; hw tanh on TRANS pipe)
#pragma unroll
        for (int v = 0; v < 8; ++v) {
            float iv = fast_sigmoid(z[0][v]);
            float fv = fast_sigmoid(z[1][v]);
            float gv = fast_tanh(z[2][v]);
            float ov = fast_sigmoid(z[3][v]);
            float c  = __builtin_fmaf(fv, cst[v], iv * gv);
            cst[v]   = c;
            float h  = ov * fast_tanh(c);
            hlast[v] = h;
            int m    = v + moff;
            h_lds[nxt][m * H_ + hcol] = f32_to_bf16(h);   // write NEXT buffer
            int brow = bt * 16 + m;
            out[((size_t)brow * T_ + t) * (2 * H_) + dir * H_ + hcol] = h;
        }
        __syncthreads();      // single barrier: nxt-buffer writes visible next step
    }

    // final (h, c): states layout = h_fw, c_fw, h_bw, c_bw each [B][H]
    float* hout = states + (size_t)dir * 2 * B_ * H_;
    float* cout = hout + (size_t)B_ * H_;
#pragma unroll
    for (int v = 0; v < 8; ++v) {
        int brow = bt * 16 + v + moff;
        hout[(size_t)brow * H_ + hcol] = hlast[v];
        cout[(size_t)brow * H_ + hcol] = cst[v];
    }
}

extern "C" void kernel_launch(void* const* d_in, const int* in_sizes, int n_in,
                              void* d_out, int out_size, void* d_ws, size_t ws_size,
                              hipStream_t stream) {
    // setup_inputs order: tokens, emb, W_fw, U_fw, b_fw, W_bw, U_bw, b_bw
    const long long* tokens = (const long long*)d_in[0];
    const float*     emb    = (const float*)d_in[1];
    const float*     W_fw   = (const float*)d_in[2];
    const float*     U_fw   = (const float*)d_in[3];
    const float*     b_fw   = (const float*)d_in[4];
    const float*     W_bw   = (const float*)d_in[5];
    const float*     U_bw   = (const float*)d_in[6];
    const float*     b_bw   = (const float*)d_in[7];
    float* out = (float*)d_out;

    // workspace carve-up (all offsets 128B-aligned)
    char* ws = (char*)d_ws;
    size_t off = 0;
    uint16_t* xt    = (uint16_t*)(ws + off); off += (size_t)T_ * B_ * E_ * 2;      // 33.5 MB
    float*    xw_fw = (float*)   (ws + off); off += (size_t)T_ * B_ * FOURH * 4;   // 134 MB
    float*    xw_bw = (float*)   (ws + off); off += (size_t)T_ * B_ * FOURH * 4;   // 134 MB
    uint16_t* Wt_fw = (uint16_t*)(ws + off); off += (size_t)FOURH * E_ * 2;
    uint16_t* Wt_bw = (uint16_t*)(ws + off); off += (size_t)FOURH * E_ * 2;
    uint16_t* Ut_fw = (uint16_t*)(ws + off); off += (size_t)FOURH * H_ * 2;
    uint16_t* Ut_bw = (uint16_t*)(ws + off); off += (size_t)FOURH * H_ * 2;

    // 0) weight transpose/convert to bf16 [N][K]
    {
        int nW = E_ * FOURH, nU = H_ * FOURH;
        transpose_cvt<<<(nW + 255) / 256, 256, 0, stream>>>(W_fw, Wt_fw, E_, FOURH);
        transpose_cvt<<<(nW + 255) / 256, 256, 0, stream>>>(W_bw, Wt_bw, E_, FOURH);
        transpose_cvt<<<(nU + 255) / 256, 256, 0, stream>>>(U_fw, Ut_fw, H_, FOURH);
        transpose_cvt<<<(nU + 255) / 256, 256, 0, stream>>>(U_bw, Ut_bw, H_, FOURH);
    }
    // 1) embedding gather -> bf16 time-major
    gather_embed<<<T_ * B_, E_, 0, stream>>>(tokens, emb, xt);

    // 2) input projection GEMMs (both directions), one 16x64 strip per wave
    {
        int strips = (T_ * B_ / 16) * (FOURH / 64);   // 4096 * 8 = 32768
        dim3 grid(strips / 8, 2);
        inproj_gemm<<<grid, 256, 0, stream>>>(xt, Wt_fw, Wt_bw, b_fw, b_bw,
                                              xw_fw, xw_bw);
    }
    // 3) sequential recurrence: 8 batch tiles x 2 directions
    {
        float* states = out + (size_t)B_ * T_ * 2 * H_;
        dim3 grid(B_ / 16, 2);
        lstm_recurrence<<<grid, 256, 0, stream>>>(xw_fw, xw_bw, Ut_fw, Ut_bw,
                                                  out, states);
    }
}
